// StackSAModuleMSGDeform_26259430048570
// MI455X (gfx1250) — matrix operand order repaired and checked by reading, not compiled, not run on hardware
//
#include <hip/hip_runtime.h>
#include <hip/hip_bf16.h>
#include <math.h>

typedef __attribute__((ext_vector_type(16))) _Float16 v16h;
typedef __attribute__((ext_vector_type(8)))  _Float16 h8;
typedef __attribute__((ext_vector_type(8)))  float    v8f;

// ---------------- problem constants (from reference setup_inputs) ----------------
#define BQ    2
#define NPTS  8192
#define CIN   32
#define RROI  64
#define GRID  64
#define MQ    (RROI * GRID)    // 4096 queries per batch
#define QTOT  (BQ * MQ)        // 8192 queries total
#define CROI  128
#define S1    16               // PRED_NSAMPLE
#define S2    32               // NSAMPLE
#define ROWS1 (QTOT * S1)      // 131072
#define ROWS2 (QTOT * S2)      // 262144
#define KPAD  64               // 35 -> padded to 64 (2 x K32 WMMA steps)
#define FCIN  (GRID * 32 + CROI) // 2176
#define BN_EPS 1e-5f

// ---------------- small utility kernels ----------------
__global__ void zero_k(float* p, int n) {
    int t = blockIdx.x * blockDim.x + threadIdx.x;
    if (t < n) p[t] = 0.f;
}

__global__ void copy_k(const float* __restrict__ src, float* __restrict__ dst, int n) {
    int t = blockIdx.x * blockDim.x + threadIdx.x;
    if (t < n) dst[t] = src[t];
}

// f32 weight (Cin x Cout) -> f16 packed directly in WMMA B-fragment layout:
//   WB[((kchunk * ntiles + coltile) * 32 + lane) * 16 + j]
//   lane l supplies B(K = kchunk*32 + (l>>4)*8 + (j&7) + (j>>3)*16 , N = coltile*16 + (l&15))
// so each lane's 16 f16 (32 bytes) are contiguous -> two aligned b128 loads in the GEMM.
__global__ void pack_weight_frag_k(const float* __restrict__ W, _Float16* __restrict__ WB,
                                   int Cin, int Cout, int Kp) {
    const int ntiles = Cout >> 4;
    const int total = (Kp >> 5) * ntiles * 32 * 16;
    int t = blockIdx.x * blockDim.x + threadIdx.x;
    if (t >= total) return;
    const int j    = t & 15;
    const int lane = (t >> 4) & 31;
    const int rest = t >> 9;
    const int coltile = rest % ntiles;
    const int kchunk  = rest / ntiles;
    const int k = kchunk * 32 + (lane >> 4) * 8 + (j & 7) + (j >> 3) * 16;
    const int c = coltile * 16 + (lane & 15);
    WB[t] = (_Float16)((k < Cin) ? W[k * Cout + c] : 0.f);
}

// ---------------- ball query: one wave32 per query, first-found order ----------------
template<int NS>
__global__ void ball_query_k(const float* __restrict__ xyz, const float* __restrict__ q,
                             const float* __restrict__ rad, float rconst,
                             int* __restrict__ idx, int* __restrict__ empty) {
    const int lane = threadIdx.x & 31;
    const int wave = threadIdx.x >> 5;
    const int qi = blockIdx.x * (blockDim.x >> 5) + wave;
    if (qi >= QTOT) return;
    const int b = qi / MQ;
    const float qx = q[qi * 3 + 0], qy = q[qi * 3 + 1], qz = q[qi * 3 + 2];
    const float r  = rad ? rad[qi] : rconst;
    const float r2 = r * r;
    const float* __restrict__ X = xyz + (size_t)b * NPTS * 3;
    int cnt = 0;
    for (int base = 0; base < NPTS && cnt < NS; base += 32) {
        const int p = base + lane;
        const float dx = X[p * 3 + 0] - qx;
        const float dy = X[p * 3 + 1] - qy;
        const float dz = X[p * 3 + 2] - qz;
        const float d2 = dx * dx + dy * dy + dz * dz;
        const bool valid = d2 < r2;
        const unsigned mask = __builtin_amdgcn_ballot_w32(valid);
        const int pre = __builtin_popcount(mask & ((1u << lane) - 1u));
        if (valid && (cnt + pre) < NS) idx[(size_t)qi * NS + cnt + pre] = p;
        cnt += __builtin_popcount(mask);  // uniform across the wave
    }
    if (cnt > NS) cnt = NS;
    if (lane == 0) {
        if (cnt == 0) {
            empty[qi] = 1;
            for (int i = 0; i < NS; ++i) idx[(size_t)qi * NS + i] = 0;
        } else {
            empty[qi] = 0;
            const int f = idx[(size_t)qi * NS];   // same-wave store->load, in order
            for (int i = cnt; i < NS; ++i) idx[(size_t)qi * NS + i] = f;
        }
    }
}

// ---------------- group/gather: build (rows x KPAD) f16 WMMA input (+ optional sigmoid weights) ----
// row built in 8 x h8 registers, stored with 8 aligned 16B stores (no 2-byte stores)
__global__ void group_k(const float* __restrict__ xyz, const float* __restrict__ feats,
                        const float* __restrict__ q, const int* __restrict__ idx,
                        const int* __restrict__ empty, _Float16* __restrict__ Xout,
                        int S, const float* __restrict__ r_q,
                        const float* __restrict__ tdecay, float* __restrict__ wout) {
    const int t = blockIdx.x * blockDim.x + threadIdx.x;
    const int total = QTOT * S;
    if (t >= total) return;
    const int qi = t / S;
    const int s  = t - qi * S;
    const int b  = qi / MQ;
    h8* row = (h8*)(Xout + (size_t)t * KPAD);
    if (empty[qi]) {
        h8 z = (h8)(_Float16)0.f;
        #pragma unroll
        for (int v = 0; v < KPAD / 8; ++v) row[v] = z;
        if (wout) wout[t] = 0.f;
        return;
    }
    const int p = idx[(size_t)qi * S + s];
    const float* X = xyz + ((size_t)b * NPTS + p) * 3;
    float rel[3];
    rel[0] = X[0] - q[qi * 3 + 0];
    rel[1] = X[1] - q[qi * 3 + 1];
    rel[2] = X[2] - q[qi * 3 + 2];
    const float* F = feats + ((size_t)b * NPTS + p) * CIN;
    #pragma unroll
    for (int v = 0; v < KPAD / 8; ++v) {
        h8 o;
        #pragma unroll
        for (int j = 0; j < 8; ++j) {
            const int e = v * 8 + j;
            float val;
            if (e < 3)            val = rel[e];
            else if (e < 3 + CIN) val = F[e - 3];
            else                  val = 0.f;
            o[j] = (_Float16)val;
        }
        row[v] = o;
    }
    if (wout) {
        const float dist = sqrtf(rel[0]*rel[0] + rel[1]*rel[1] + rel[2]*rel[2]);
        const float x = (r_q[qi] - dist) / (1.0f * tdecay[0]);  // TEMPERATURE = 1
        wout[t] = 1.f / (1.f + expf(-x));
    }
}

// ---------------- WMMA GEMM: Y(rows x Cout) = X(rows x Kp) @ WB(frag-packed), + BN stats ----------
// one wave per 16x16 output tile; 4 waves per block along rows.
// A: two aligned b128 loads/lane ; B: two aligned b128 loads/lane (frag-packed, L2-resident)
__global__ void wmma_gemm_k(const _Float16* __restrict__ X, const _Float16* __restrict__ WB,
                            float* __restrict__ Y, float* __restrict__ stats,
                            int Kp, int Cout) {
    const int lane = threadIdx.x & 31;
    const int wave = threadIdx.x >> 5;
    const int rbase = (blockIdx.x * 4 + wave) * 16;
    const int cbase = blockIdx.y * 16;
    const int ntiles = gridDim.y;
    const int m  = lane & 15;   // A row / D column
    const int hi = lane >> 4;   // K-half selector
    v8f acc = {};
    for (int kc = 0; kc < Kp; kc += 32) {
        // A fragment: 16x32 f16 (lane<16 -> K 0-7,16-23 ; lane>=16 -> K 8-15,24-31)
        const h8* ap = (const h8*)(X + (size_t)(rbase + m) * Kp + kc);
        const h8 a0 = ap[hi];
        const h8 a1 = ap[2 + hi];
        const v16h a = __builtin_shufflevector(a0, a1,
            0, 1, 2, 3, 4, 5, 6, 7, 8, 9, 10, 11, 12, 13, 14, 15);
        // B fragment: pre-packed per (kchunk, coltile, lane): 32 contiguous bytes
        const h8* bp = (const h8*)(WB + ((size_t)((kc >> 5) * ntiles + blockIdx.y) * 32 + lane) * 16);
        const h8 b0 = bp[0];
        const h8 b1 = bp[1];
        const v16h bf = __builtin_shufflevector(b0, b1,
            0, 1, 2, 3, 4, 5, 6, 7, 8, 9, 10, 11, 12, 13, 14, 15);
        acc = __builtin_amdgcn_wmma_f32_16x16x32_f16(
            /*neg_a=*/false, a, /*neg_b=*/false, bf,
            /*c_mod=*/(short)0, acc, /*reuse_a=*/false, /*reuse_b=*/false);
    }
    // D layout: lane<16 -> col=lane, rows v ; lane>=16 -> col=lane-16, rows v+8
    float s = 0.f, sq = 0.f;
    #pragma unroll
    for (int v = 0; v < 8; ++v) {
        const int mm = v + hi * 8;
        const float val = acc[v];
        Y[(size_t)(rbase + mm) * Cout + cbase + m] = val;
        s += val; sq += val * val;
    }
    if (stats) {
        atomicAdd(&stats[cbase + m], s);
        atomicAdd(&stats[Cout + cbase + m], sq);
    }
}

// ---------------- BN(training stats) + ReLU -> next-layer f16 input (8 channels/thread) ----------
__global__ void bn_relu_h_k(const float* __restrict__ Y, const float* __restrict__ stats,
                            const float* __restrict__ g, const float* __restrict__ bta,
                            _Float16* __restrict__ Xn, int total8, int Cout, float invn) {
    const int t = blockIdx.x * blockDim.x + threadIdx.x;   // one per 8 consecutive elements
    if (t >= total8) return;
    const int e0 = t * 8;
    const int c0 = e0 % Cout;   // Cout multiple of 8, so 8 consecutive channels
    h8 o;
    #pragma unroll
    for (int j = 0; j < 8; ++j) {
        const int c = c0 + j;
        const float mu  = stats[c] * invn;
        const float inv = rsqrtf(stats[Cout + c] * invn - mu * mu + BN_EPS);
        const float v = (Y[e0 + j] - mu) * inv * g[c] + bta[c];
        o[j] = (_Float16)fmaxf(v, 0.f);
    }
    *(h8*)(Xn + e0) = o;
}

// ---------------- fused BN + ReLU + max over samples (pred path) ----------------
__global__ void maxpool_bn_k(const float* __restrict__ Y, const float* __restrict__ stats,
                             const float* __restrict__ g, const float* __restrict__ bta,
                             float* __restrict__ pooled, float invn) {
    const int t = blockIdx.x * blockDim.x + threadIdx.x;  // qi*32 + c
    if (t >= QTOT * 32) return;
    const int qi = t >> 5, c = t & 31;
    const float mu  = stats[c] * invn;
    const float inv = rsqrtf(stats[32 + c] * invn - mu * mu + BN_EPS);
    float mx = -1e30f;
    for (int s = 0; s < S1; ++s) {
        float v = (Y[(size_t)(qi * S1 + s) * 32 + c] - mu) * inv * g[c] + bta[c];
        mx = fmaxf(mx, fmaxf(v, 0.f));
    }
    pooled[t] = mx;
}

// ---------------- FC (2176 -> 1) per ROI, radius clamp, broadcast to per-query radii ----------
__global__ void fc_radius_k(const float* __restrict__ pooled, const float* __restrict__ roif,
                            const float* __restrict__ fcw, const float* __restrict__ fcb,
                            float* __restrict__ r_q) {
    __shared__ float red[256];
    const int roi = blockIdx.x;              // 0 .. B*R-1
    const int b = roi / RROI, r = roi - b * RROI;
    float acc = 0.f;
    for (int j = threadIdx.x; j < FCIN; j += blockDim.x) {
        float x;
        if (j < GRID * 32) {
            const int gi = j >> 5, c = j & 31;
            x = pooled[(size_t)(b * MQ + r * GRID + gi) * 32 + c];
        } else {
            x = roif[(size_t)roi * CROI + (j - GRID * 32)];
        }
        acc += x * fcw[j];
    }
    red[threadIdx.x] = acc;
    __syncthreads();
    for (int st = 128; st > 0; st >>= 1) {
        if ((int)threadIdx.x < st) red[threadIdx.x] += red[threadIdx.x + st];
        __syncthreads();
    }
    if (threadIdx.x == 0)
        red[0] = fmaxf((red[0] + fcb[0]) / 10.0f + 0.8f, 0.01f);  // DIV_COEF, RADIUS, MIN_R
    __syncthreads();
    const float rv = red[0];
    for (int gi = threadIdx.x; gi < GRID; gi += blockDim.x)
        r_q[(size_t)b * MQ + r * GRID + gi] = rv;
}

// ---------------- fused BN + ReLU + weighted max over samples (deform path) -> d_out ------------
__global__ void final_wmax_k(const float* __restrict__ Y, const float* __restrict__ stats,
                             const float* __restrict__ g, const float* __restrict__ bta,
                             const float* __restrict__ w, float* __restrict__ out, float invn) {
    const int t = blockIdx.x * blockDim.x + threadIdx.x;  // qi*64 + c
    if (t >= QTOT * 64) return;
    const int qi = t >> 6, c = t & 63;
    const float mu  = stats[c] * invn;
    const float inv = rsqrtf(stats[64 + c] * invn - mu * mu + BN_EPS);
    float mx = -1e30f;
    for (int s = 0; s < S2; ++s) {
        float v = (Y[(size_t)(qi * S2 + s) * 64 + c] - mu) * inv * g[c] + bta[c];
        v = fmaxf(v, 0.f) * w[qi * S2 + s];
        mx = fmaxf(mx, v);
    }
    out[t] = mx;
}

// ---------------- host launch ----------------
extern "C" void kernel_launch(void* const* d_in, const int* in_sizes, int n_in,
                              void* d_out, int out_size, void* d_ws, size_t ws_size,
                              hipStream_t stream) {
    const float* xyz    = (const float*)d_in[0];   // (B,N,3)
    const float* feats  = (const float*)d_in[1];   // (B,N,32)
    const float* rois   = (const float*)d_in[2];   // (B,R,G,3) == new_xyz (B*M,3)
    const float* roif   = (const float*)d_in[3];   // (B,R,128)
    const float* tdecay = (const float*)d_in[4];   // (1,)
    const float* pw0 = (const float*)d_in[5];
    const float* pg0 = (const float*)d_in[6];
    const float* pb0 = (const float*)d_in[7];
    const float* pw1 = (const float*)d_in[8];
    const float* pg1 = (const float*)d_in[9];
    const float* pb1 = (const float*)d_in[10];
    const float* fw0 = (const float*)d_in[11];
    const float* fg0 = (const float*)d_in[12];
    const float* fb0 = (const float*)d_in[13];
    const float* fw1 = (const float*)d_in[14];
    const float* fg1 = (const float*)d_in[15];
    const float* fb1 = (const float*)d_in[16];
    const float* fcw = (const float*)d_in[17];
    const float* fcb = (const float*)d_in[18];
    float* out = (float*)d_out;  // [0,24576): new_xyz ; [24576, +QTOT*64): new_feats

    // ---- workspace carve-up (256B aligned) ----
    char* base = (char*)d_ws;
    size_t off = 0;
    auto carve = [&](size_t bytes) -> char* {
        char* p = base + off;
        off += (bytes + 255) & ~(size_t)255;
        return p;
    };
    _Float16* XA    = (_Float16*)carve((size_t)ROWS2 * KPAD * sizeof(_Float16)); // group output (both paths)
    _Float16* XB    = (_Float16*)carve((size_t)ROWS2 * KPAD * sizeof(_Float16)); // layer-2 inputs
    float*    YRAW  = (float*)   carve((size_t)ROWS2 * 64  * sizeof(float));     // raw GEMM out (reused)
    int*      idxp  = (int*)     carve((size_t)QTOT * S1 * sizeof(int));
    int*      idxd  = (int*)     carve((size_t)QTOT * S2 * sizeof(int));
    int*      emp_p = (int*)     carve((size_t)QTOT * sizeof(int));
    int*      emp_d = (int*)     carve((size_t)QTOT * sizeof(int));
    float*    pooled= (float*)   carve((size_t)QTOT * 32 * sizeof(float));
    float*    r_q   = (float*)   carve((size_t)QTOT * sizeof(float));
    float*    wbuf  = (float*)   carve((size_t)QTOT * S2 * sizeof(float));
    float*    stats = (float*)   carve(4 * 128 * sizeof(float));  // 4 layers x (sum|sumsq) x <=64ch
    _Float16* Wp0h  = (_Float16*)carve(64 * 32 * sizeof(_Float16));
    _Float16* Wp1h  = (_Float16*)carve(32 * 32 * sizeof(_Float16));
    _Float16* Wf0h  = (_Float16*)carve(64 * 64 * sizeof(_Float16));
    _Float16* Wf1h  = (_Float16*)carve(64 * 64 * sizeof(_Float16));
    float* st_p0 = stats, *st_p1 = stats + 128, *st_f0 = stats + 256, *st_f1 = stats + 384;

    const int T = 256;
    // zero BN stats every call (harness does not re-poison between replays)
    zero_k<<<(512 + T - 1) / T, T, 0, stream>>>(stats, 512);
    // output 0: new_xyz = rois flattened
    copy_k<<<(QTOT * 3 + T - 1) / T, T, 0, stream>>>(rois, out, QTOT * 3);
    // pack f16 weights into WMMA B-fragment layout (pad Cin 35 -> 64)
    pack_weight_frag_k<<<(64 * 32 + T - 1) / T, T, 0, stream>>>(pw0, Wp0h, 35, 32, 64);
    pack_weight_frag_k<<<(32 * 32 + T - 1) / T, T, 0, stream>>>(pw1, Wp1h, 32, 32, 32);
    pack_weight_frag_k<<<(64 * 64 + T - 1) / T, T, 0, stream>>>(fw0, Wf0h, 35, 64, 64);
    pack_weight_frag_k<<<(64 * 64 + T - 1) / T, T, 0, stream>>>(fw1, Wf1h, 64, 64, 64);

    // ---- pred path ----
    ball_query_k<S1><<<QTOT / 8, 256, 0, stream>>>(xyz, rois, nullptr, 0.8f, idxp, emp_p);
    group_k<<<(ROWS1 + T - 1) / T, T, 0, stream>>>(xyz, feats, rois, idxp, emp_p, XA,
                                                   S1, nullptr, nullptr, nullptr);
    wmma_gemm_k<<<dim3(ROWS1 / 64, 32 / 16), 128, 0, stream>>>(XA, Wp0h, YRAW, st_p0, KPAD, 32);
    bn_relu_h_k<<<(ROWS1 * 32 / 8 + T - 1) / T, T, 0, stream>>>(YRAW, st_p0, pg0, pb0, XB,
                                                                ROWS1 * 32 / 8, 32, 1.f / ROWS1);
    wmma_gemm_k<<<dim3(ROWS1 / 64, 32 / 16), 128, 0, stream>>>(XB, Wp1h, YRAW, st_p1, 32, 32);
    maxpool_bn_k<<<(QTOT * 32 + T - 1) / T, T, 0, stream>>>(YRAW, st_p1, pg1, pb1, pooled, 1.f / ROWS1);
    fc_radius_k<<<BQ * RROI, 256, 0, stream>>>(pooled, roif, fcw, fcb, r_q);

    // ---- deform path ----
    ball_query_k<S2><<<QTOT / 8, 256, 0, stream>>>(xyz, rois, r_q, 0.f, idxd, emp_d);
    group_k<<<(ROWS2 + T - 1) / T, T, 0, stream>>>(xyz, feats, rois, idxd, emp_d, XA,
                                                   S2, r_q, tdecay, wbuf);
    wmma_gemm_k<<<dim3(ROWS2 / 64, 64 / 16), 128, 0, stream>>>(XA, Wf0h, YRAW, st_f0, KPAD, 64);
    bn_relu_h_k<<<(ROWS2 * 64 / 8 + T - 1) / T, T, 0, stream>>>(YRAW, st_f0, fg0, fb0, XB,
                                                                ROWS2 * 64 / 8, 64, 1.f / ROWS2);
    wmma_gemm_k<<<dim3(ROWS2 / 64, 64 / 16), 128, 0, stream>>>(XB, Wf1h, YRAW, st_f1, 64, 64);
    final_wmax_k<<<(QTOT * 64 + T - 1) / T, T, 0, stream>>>(YRAW, st_f1, fg1, fb1, wbuf,
                                                            out + QTOT * 3, 1.f / ROWS2);
    (void)in_sizes; (void)n_in; (void)out_size; (void)ws_size;
}
